// WorldModel_48507360641704
// MI455X (gfx1250) — matrix operand order repaired and checked by reading, not compiled, not run on hardware
//
#include <hip/hip_runtime.h>
#include <hip/hip_bf16.h>

#define B_   16
#define L_   512
#define S_   64
#define H_   1024
#define E_   1024
#define DI_  2048
#define NS_  4
#define BL_  (B_ * L_)        /* 8192 */
#define SZ_  (B_ * L_ * S_)   /* 524288 */

typedef __attribute__((ext_vector_type(16))) __bf16 v16bf;
typedef __attribute__((ext_vector_type(8)))  float  v8f;

template<int ACT>
__device__ __forceinline__ float apply_act(float x) {
    if constexpr (ACT == 1) {            // ELU (alpha=1)
        return x > 0.f ? x : expm1f(x);
    } else if constexpr (ACT == 2) {     // softplus
        return x > 30.f ? x : log1pf(expf(x));
    } else {
        return x;
    }
}

__device__ __forceinline__ int imin(int a, int b) { return a < b ? a : b; }

// ---------------------------------------------------------------------------
// Weight pre-pack: f32 (Nr x Kc, row stride ldsrc, col offset koff) ->
// zero-padded bf16 hi plane [Np*Kp] followed by lo plane [Np*Kp].
// Np % 128 == 0, Kp % 32 == 0, so GEMM B-fragment loads need no bounds logic.
// ---------------------------------------------------------------------------
__global__ __launch_bounds__(256)
void pack_w_k(const float* __restrict__ W, int Nr, int Kc, int ldsrc, int koff,
              __bf16* __restrict__ dst, int Np, int Kp)
{
    long i = (long)blockIdx.x * 256 + threadIdx.x;
    long total = (long)Np * Kp;
    if (i >= total) return;
    int k = (int)(i % Kp);
    int n = (int)(i / Kp);
    float v = (n < Nr && k < Kc) ? W[(long)n * ldsrc + koff + k] : 0.f;
    __bf16 h = (__bf16)v;
    dst[i] = h;
    dst[total + i] = (__bf16)(v - (float)h);
}

// ---------------------------------------------------------------------------
// Split-precision WMMA GEMM:  C = act( A(MxK) * W(NxK)^T + bias [+C] )
// A is f32 (hi/lo decomposed in-kernel, amortized over 8 N-tiles); W comes
// pre-packed as padded bf16 hi/lo planes (row stride ldw=Kp), loaded as
// v16bf fragments from two aligned 16B loads with zero bounds logic.  3 WMMAs per tile pair
// (hi*hi + hi*lo + lo*hi) give ~fp32 accuracy on v_wmma_f32_16x16x32_bf16.
// WMMAs are issued grouped by round across the 8 N-tiles so consecutive
// WMMAs never share an accumulator (no D->C hazard stalls).
// Block = 256 threads = 8 waves. Wave tile = 16(M) x 128(N). Block = 128x128.
// ---------------------------------------------------------------------------
template<int ACT, bool ADD_C, bool HAS_BIAS, bool FULL>
__global__ __launch_bounds__(256)
void wmma_gemm(const float* __restrict__ A, int lda,
               const __bf16* __restrict__ WH, const __bf16* __restrict__ WL,
               int ldw,
               const float* __restrict__ bias,
               float* __restrict__ C, int ldc,
               int M, int N, int K)
{
    const int lane = threadIdx.x & 31;
    const int wave = threadIdx.x >> 5;
    const int lm   = lane & 15;
    const int half = lane >> 4;
    const int m0   = blockIdx.x * 128 + wave * 16;
    const int n0   = blockIdx.y * 128;

    v8f acc[8];
#pragma unroll
    for (int t = 0; t < 8; ++t) acc[t] = (v8f){0.f,0.f,0.f,0.f,0.f,0.f,0.f,0.f};

    const int  row    = m0 + lm;
    const bool row_ok = FULL ? true : (row < M);
    const float* Arow = A + (long)(FULL ? row : imin(row, M - 1)) * (long)lda;

    const __bf16* whrow[8];
    const __bf16* wlrow[8];
#pragma unroll
    for (int t = 0; t < 8; ++t) {
        const int n = n0 + 16 * t + lm;           // always < Np (padded)
        whrow[t] = WH + (long)n * ldw;
        wlrow[t] = WL + (long)n * ldw;
    }

    for (int k0 = 0; k0 < K; k0 += 32) {
        // ---- A fragment (16x32, ISA bf16 A-layout), hi/lo decomposition ----
        float av[16];
        if constexpr (FULL) {
            const float* pa = Arow + k0 + 8 * half;
            float4 q0 = *reinterpret_cast<const float4*>(pa);
            float4 q1 = *reinterpret_cast<const float4*>(pa + 4);
            float4 q2 = *reinterpret_cast<const float4*>(pa + 16);
            float4 q3 = *reinterpret_cast<const float4*>(pa + 20);
            av[0]=q0.x; av[1]=q0.y; av[2]=q0.z; av[3]=q0.w;
            av[4]=q1.x; av[5]=q1.y; av[6]=q1.z; av[7]=q1.w;
            av[8]=q2.x; av[9]=q2.y; av[10]=q2.z; av[11]=q2.w;
            av[12]=q3.x; av[13]=q3.y; av[14]=q3.z; av[15]=q3.w;
        } else {
#pragma unroll
            for (int p = 0; p < 8; ++p) {
                const int kk = k0 + (p < 4 ? 2 * p : 16 + 2 * (p - 4)) + 8 * half;
                float2 t2 = *reinterpret_cast<const float2*>(Arow + imin(kk, K - 2));
                const bool ok = row_ok && (kk < K);
                av[2 * p]     = ok ? t2.x : 0.f;
                av[2 * p + 1] = ok ? t2.y : 0.f;
            }
        }
        v16bf a_hi, a_lo;
#pragma unroll
        for (int e = 0; e < 16; ++e) {
            __bf16 h = (__bf16)av[e];
            a_hi[e] = h;
            a_lo[e] = (__bf16)(av[e] - (float)h);
        }

        // ---- B fragments: aligned 16B half-loads from pre-packed planes ----
        v16bf bh[8], bl[8];
#pragma unroll
        for (int t = 0; t < 8; ++t) {
            typedef __attribute__((ext_vector_type(8))) __bf16 v8bf_;
            const long off = k0 + 8 * half;
            v8bf_ h0 = *reinterpret_cast<const v8bf_*>(whrow[t] + off);
            v8bf_ h1 = *reinterpret_cast<const v8bf_*>(whrow[t] + off + 16);
            v8bf_ l0 = *reinterpret_cast<const v8bf_*>(wlrow[t] + off);
            v8bf_ l1 = *reinterpret_cast<const v8bf_*>(wlrow[t] + off + 16);
            bh[t] = __builtin_shufflevector(h0, h1, 0,1,2,3,4,5,6,7,8,9,10,11,12,13,14,15);
            bl[t] = __builtin_shufflevector(l0, l1, 0,1,2,3,4,5,6,7,8,9,10,11,12,13,14,15);
        }

        // ---- 24 WMMAs, grouped by round: no back-to-back same-acc dep ----
#pragma unroll
        for (int t = 0; t < 8; ++t)
            acc[t] = __builtin_amdgcn_wmma_f32_16x16x32_bf16(
                         false, a_hi, false, bh[t], (short)0, acc[t], false, false);
#pragma unroll
        for (int t = 0; t < 8; ++t)
            acc[t] = __builtin_amdgcn_wmma_f32_16x16x32_bf16(
                         false, a_hi, false, bl[t], (short)0, acc[t], false, false);
#pragma unroll
        for (int t = 0; t < 8; ++t)
            acc[t] = __builtin_amdgcn_wmma_f32_16x16x32_bf16(
                         false, a_lo, false, bh[t], (short)0, acc[t], false, false);
    }

    // ---- store C (f32 16x16 C/D layout: lane -> N, vgpr -> M) ----
#pragma unroll
    for (int t = 0; t < 8; ++t) {
        const int n = n0 + 16 * t + lm;
        if (!FULL && n >= N) continue;
        float bv = 0.f;
        if constexpr (HAS_BIAS) bv = bias[n];
#pragma unroll
        for (int v = 0; v < 8; ++v) {
            const int m = m0 + v + 8 * half;
            if (!FULL && m >= M) continue;
            float c = acc[t][v] + bv;
            if constexpr (ADD_C) c += C[(long)m * ldc + n];
            C[(long)m * ldc + n] = apply_act<ACT>(c);
        }
    }
}

// ---------------------------------------------------------------------------
// x[b,l,:] = Wa_act[b,l,:] + (l==0 ? s0_emb[b,:] : eo[b,l-1,:])   (in place)
// ---------------------------------------------------------------------------
__global__ __launch_bounds__(256)
void shift_add_k(float* __restrict__ X, const float* __restrict__ EO,
                 const float* __restrict__ S0E)
{
    long i = (long)blockIdx.x * 256 + threadIdx.x;   // BL_*H_
    int  hcol = (int)(i % H_);
    long r = i / H_;
    int  l = (int)(r % L_);
    long b = r / L_;
    float add = (l == 0) ? S0E[b * H_ + hcol] : EO[(r - 1) * H_ + hcol];
    X[i] += add;
}

// ---------------------------------------------------------------------------
// Depthwise causal conv (K=4) + bias + SiLU.  xm = first DI_ columns of the
// in_proj output (row stride 4096).
// ---------------------------------------------------------------------------
__global__ __launch_bounds__(256)
void conv_silu_k(const float* __restrict__ IP, const float* __restrict__ cw,
                 const float* __restrict__ cb, float* __restrict__ U)
{
    long i = (long)blockIdx.x * 256 + threadIdx.x;   // BL_*DI_
    int  d = (int)(i % DI_);
    long r = i / DI_;                                 // b*L + l
    int  l = (int)(r % L_);
    float acc = cb[d];
#pragma unroll
    for (int j = 0; j < 4; ++j) {
        int lj = l - 3 + j;
        if (lj >= 0) acc += IP[(r - (3 - j)) * 4096 + d] * cw[d * 4 + j];
    }
    U[i] = acc / (1.f + expf(-acc));
}

// ---------------------------------------------------------------------------
// Selective scan over L.  One thread per (b,d) channel: 32768 parallel chains.
// ---------------------------------------------------------------------------
__global__ __launch_bounds__(256)
void scan_k(const float* __restrict__ DELTA, const float* __restrict__ U,
            const float* __restrict__ IP,    const float* __restrict__ DBL,
            const float* __restrict__ A_log, const float* __restrict__ Dv,
            float* __restrict__ Y)
{
    int g = blockIdx.x * 256 + threadIdx.x;          // b*DI_ + d
    int d = g % DI_;
    int b = g / DI_;
    float An[NS_];
#pragma unroll
    for (int n = 0; n < NS_; ++n) An[n] = -expf(A_log[d * NS_ + n]);
    float Dd = Dv[d];
    float h[NS_] = {0.f, 0.f, 0.f, 0.f};
    for (int t = 0; t < L_; ++t) {
        long  r  = (long)b * L_ + t;
        float dl = DELTA[r * DI_ + d];
        float u  = U[r * DI_ + d];
        float z  = IP[r * 4096 + 2048 + d];
        const float* bl = DBL + r * 72;
        float y = 0.f;
#pragma unroll
        for (int n = 0; n < NS_; ++n) {
            h[n] = expf(dl * An[n]) * h[n] + dl * bl[64 + n] * u;
            y += h[n] * bl[68 + n];
        }
        y += u * Dd;
        Y[r * DI_ + d] = y * (z / (1.f + expf(-z)));
    }
}

// ---------------------------------------------------------------------------
// Row LayerNorm over H_=1024, one block per row.
// ---------------------------------------------------------------------------
__global__ __launch_bounds__(256)
void layernorm_k(const float* __restrict__ Xin, const float* __restrict__ g,
                 const float* __restrict__ be, float* __restrict__ Out)
{
    __shared__ float red[256];
    const long  row = blockIdx.x;
    const float* xr = Xin + row * H_;
    float s = 0.f, s2 = 0.f;
    for (int i = threadIdx.x; i < H_; i += 256) { float v = xr[i]; s += v; s2 += v * v; }
    red[threadIdx.x] = s; __syncthreads();
    for (int o = 128; o > 0; o >>= 1) {
        if (threadIdx.x < o) red[threadIdx.x] += red[threadIdx.x + o];
        __syncthreads();
    }
    float mean = red[0] * (1.f / H_); __syncthreads();
    red[threadIdx.x] = s2; __syncthreads();
    for (int o = 128; o > 0; o >>= 1) {
        if (threadIdx.x < o) red[threadIdx.x] += red[threadIdx.x + o];
        __syncthreads();
    }
    float var = red[0] * (1.f / H_) - mean * mean;
    float inv = rsqrtf(var + 1e-5f);
    for (int i = threadIdx.x; i < H_; i += 256)
        Out[row * H_ + i] = (xr[i] - mean) * inv * g[i] + be[i];
}

// ---------------------------------------------------------------------------
// Split prior/post into (mu, softplus(log_std)) and apply reparameterization.
// ---------------------------------------------------------------------------
__global__ __launch_bounds__(256)
void finalize_k(const float* __restrict__ PRI, const float* __restrict__ POS,
                const float* __restrict__ NP, const float* __restrict__ NQ,
                float* __restrict__ out)
{
    long i = (long)blockIdx.x * 256 + threadIdx.x;   // SZ_
    long r = i / S_;
    int  j = (int)(i % S_);
    float mup = PRI[r * 128 + j];
    float lsp = PRI[r * 128 + 64 + j];
    float sp  = lsp > 30.f ? lsp : log1pf(expf(lsp));
    float muq = POS[r * 128 + j];
    float lsq = POS[r * 128 + 64 + j];
    float sq  = lsq > 30.f ? lsq : log1pf(expf(lsq));
    out[i]                  = mup;
    out[(long)SZ_ * 1 + i]  = sp;
    out[(long)SZ_ * 2 + i]  = muq;
    out[(long)SZ_ * 3 + i]  = sq;
    out[(long)SZ_ * 4 + i]  = mup + NP[i] * sp;
    out[(long)SZ_ * 5 + i]  = muq + NQ[i] * sq;
}

// ---------------------------------------------------------------------------
struct PW { const __bf16* h; const __bf16* l; int kp; };

static PW pack_weight(hipStream_t s, float*& cur, const float* W,
                      int Nr, int Kc, int ldsrc, int koff)
{
    int Np = (Nr + 127) & ~127;
    int Kp = (Kc + 31) & ~31;
    __bf16* dst = (__bf16*)cur;
    long total = (long)Np * Kp;
    pack_w_k<<<(int)((total + 255) / 256), 256, 0, s>>>(W, Nr, Kc, ldsrc, koff,
                                                        dst, Np, Kp);
    cur += total;   // hi plane + lo plane = 4 bytes/element = 1 float slot
    return { dst, dst + total, Kp };
}

static void launch_gemm(hipStream_t s, const float* A, int lda, const PW& w,
                        const float* bias, float* C, int ldc,
                        int M, int N, int K, int actm, bool addc)
{
    dim3 grid((M + 127) / 128, (N + 127) / 128);
    dim3 block(256);
    const bool full = !addc && (M % 128 == 0) && (N % 128 == 0) && (K % 32 == 0);
    if (full) {
        if (actm == 1)
            (wmma_gemm<1, false, true , true>)<<<grid, block, 0, s>>>(A, lda, w.h, w.l, w.kp, bias, C, ldc, M, N, K);
        else if (actm == 2)
            (wmma_gemm<2, false, true , true>)<<<grid, block, 0, s>>>(A, lda, w.h, w.l, w.kp, bias, C, ldc, M, N, K);
        else if (bias)
            (wmma_gemm<0, false, true , true>)<<<grid, block, 0, s>>>(A, lda, w.h, w.l, w.kp, bias, C, ldc, M, N, K);
        else
            (wmma_gemm<0, false, false, true>)<<<grid, block, 0, s>>>(A, lda, w.h, w.l, w.kp, bias, C, ldc, M, N, K);
    } else {
        if (addc)
            (wmma_gemm<1, true , true , false>)<<<grid, block, 0, s>>>(A, lda, w.h, w.l, w.kp, bias, C, ldc, M, N, K);
        else if (actm == 1)
            (wmma_gemm<1, false, true , false>)<<<grid, block, 0, s>>>(A, lda, w.h, w.l, w.kp, bias, C, ldc, M, N, K);
        else if (actm == 2)
            (wmma_gemm<2, false, true , false>)<<<grid, block, 0, s>>>(A, lda, w.h, w.l, w.kp, bias, C, ldc, M, N, K);
        else if (bias)
            (wmma_gemm<0, false, true , false>)<<<grid, block, 0, s>>>(A, lda, w.h, w.l, w.kp, bias, C, ldc, M, N, K);
        else
            (wmma_gemm<0, false, false, false>)<<<grid, block, 0, s>>>(A, lda, w.h, w.l, w.kp, bias, C, ldc, M, N, K);
    }
}

extern "C" void kernel_launch(void* const* d_in, const int* in_sizes, int n_in,
                              void* d_out, int out_size, void* d_ws, size_t ws_size,
                              hipStream_t stream)
{
    const float* actions   = (const float*)d_in[0];
    const float* s0        = (const float*)d_in[1];
    const float* enc_o     = (const float*)d_in[2];
    const float* noise_p   = (const float*)d_in[3];
    const float* noise_q   = (const float*)d_in[4];
    const float* Wa_w      = (const float*)d_in[5];
    const float* Wa_b      = (const float*)d_in[6];
    const float* Ws_w      = (const float*)d_in[7];
    const float* Ws_b      = (const float*)d_in[8];
    const float* We_w      = (const float*)d_in[9];
    const float* We_b      = (const float*)d_in[10];
    const float* in_proj_w = (const float*)d_in[11];
    const float* conv_w    = (const float*)d_in[12];
    const float* conv_b    = (const float*)d_in[13];
    const float* x_proj_w  = (const float*)d_in[14];
    const float* dt_proj_w = (const float*)d_in[15];
    const float* dt_proj_b = (const float*)d_in[16];
    const float* A_log     = (const float*)d_in[17];
    const float* Dvec      = (const float*)d_in[18];
    const float* out_projw = (const float*)d_in[19];
    const float* gamma_h   = (const float*)d_in[20];
    const float* beta_h    = (const float*)d_in[21];
    const float* p_w1 = (const float*)d_in[22]; const float* p_b1 = (const float*)d_in[23];
    const float* p_w2 = (const float*)d_in[24]; const float* p_b2 = (const float*)d_in[25];
    const float* p_w3 = (const float*)d_in[26]; const float* p_b3 = (const float*)d_in[27];
    const float* q_w1 = (const float*)d_in[28]; const float* q_b1 = (const float*)d_in[29];
    const float* q_w2 = (const float*)d_in[30]; const float* q_b2 = (const float*)d_in[31];
    const float* q_w3 = (const float*)d_in[32]; const float* q_b3 = (const float*)d_in[33];

    // Workspace layout (floats).  Regions aliased by dependency order:
    //   [0 .. 16.7M)  : X (8.4M) + EO (8.4M)  -> DELTA (16.7M) -> H1PRE (8.4M)
    //   [16.7M ..)    : S0E, IP, U (later T1/T2/PRI/POS), DBL, Y
    //   [84.5M ..)    : packed bf16 hi/lo weight planes (launch-invariant)
    float* ws    = (float*)d_ws;
    float* X     = ws + 0;
    float* EO    = ws + 8388608;
    float* DELTA = ws + 0;
    float* H1P   = ws + 0;
    float* S0E   = ws + 16777216;
    float* IP    = ws + 16793600;
    float* U     = ws + 50348032;
    float* T1    = ws + 50348032;
    float* T2    = ws + 51986432;
    float* PRI   = ws + 53624832;
    float* POS   = ws + 54673408;
    float* DBL   = ws + 67125248;
    float* Y     = ws + 67715072;
    float* cur   = ws + 84492288;       // packed weights grow from here

    float* out = (float*)d_out;
    float* H1  = out + (long)SZ_ * 6;   // H1 is the 7th output

    // 0) pack all weights to padded bf16 hi/lo planes (one streaming pass)
    PW pWe  = pack_weight(stream, cur, We_w,      H_,   E_,   E_,   0);
    PW pWa  = pack_weight(stream, cur, Wa_w,      H_,   32,   32,   0);
    PW pWs  = pack_weight(stream, cur, Ws_w,      H_,   S_,   S_,   0);
    PW pIn  = pack_weight(stream, cur, in_proj_w, 4096, H_,   H_,   0);
    PW pXp  = pack_weight(stream, cur, x_proj_w,  72,   DI_,  DI_,  0);
    PW pDt  = pack_weight(stream, cur, dt_proj_w, DI_,  64,   64,   0);
    PW pOut = pack_weight(stream, cur, out_projw, H_,   DI_,  DI_,  0);
    PW pP1  = pack_weight(stream, cur, p_w1,      200,  H_,   H_,   0);
    PW pP2  = pack_weight(stream, cur, p_w2,      200,  200,  200,  0);
    PW pP3  = pack_weight(stream, cur, p_w3,      128,  200,  200,  0);
    PW pQ1a = pack_weight(stream, cur, q_w1,      200,  H_,   2048, 0);
    PW pQ1b = pack_weight(stream, cur, q_w1,      200,  E_,   2048, 1024);
    PW pQ2  = pack_weight(stream, cur, q_w2,      200,  200,  200,  0);
    PW pQ3  = pack_weight(stream, cur, q_w3,      128,  200,  200,  0);

    // 1) eo = elu(enc_o @ We^T + b)                               [FULL]
    launch_gemm(stream, enc_o, E_, pWe, We_b, EO, H_, BL_, H_, E_, 1, false);
    // 2) Wa_act = elu(actions @ Wa^T + b) -> X                    [FULL]
    launch_gemm(stream, actions, 32, pWa, Wa_b, X, H_, BL_, H_, 32, 1, false);
    // 3) s0_emb = elu(s0 @ Ws^T + b)                              [edge: M=16]
    launch_gemm(stream, s0, S_, pWs, Ws_b, S0E, H_, B_, H_, S_, 1, false);
    // 4) x = Wa_act + shifted(eo, s0_emb)
    shift_add_k<<<(BL_ * H_) / 256, 256, 0, stream>>>(X, EO, S0E);
    // 5) in_proj: (8192x1024) @ (4096x1024)^T -> IP (xm | z)      [FULL]
    launch_gemm(stream, X, H_, pIn, nullptr, IP, 4096, BL_, 4096, H_, 0, false);
    // 6) u = silu(depthwise_conv(xm) + conv_b)
    conv_silu_k<<<(BL_ * DI_) / 256, 256, 0, stream>>>(IP, conv_w, conv_b, U);
    // 7) dbl = u @ x_proj^T  (N=72)                               [edge: N]
    launch_gemm(stream, U, DI_, pXp, nullptr, DBL, 72, BL_, 72, DI_, 0, false);
    // 8) delta = softplus(dbl[:, :64] @ dt_proj^T + b)            [FULL]
    launch_gemm(stream, DBL, 72, pDt, dt_proj_b, DELTA, DI_, BL_, DI_, 64, 2, false);
    // 9) selective scan + D-skip + silu(z) gate -> Y
    scan_k<<<(B_ * DI_) / 256, 256, 0, stream>>>(DELTA, U, IP, DBL, A_log, Dvec, Y);
    // 10) out_proj                                                [FULL]
    launch_gemm(stream, Y, DI_, pOut, nullptr, H1P, H_, BL_, H_, DI_, 0, false);
    // 11) LayerNorm -> H1 (written directly into d_out)
    layernorm_k<<<BL_, 256, 0, stream>>>(H1P, gamma_h, beta_h, H1);
    // 12-14) prior MLP                                            [edges]
    launch_gemm(stream, H1, H_, pP1, p_b1, T1, 200, BL_, 200, H_, 1, false);
    launch_gemm(stream, T1, 200, pP2, p_b2, T2, 200, BL_, 200, 200, 1, false);
    launch_gemm(stream, T2, 200, pP3, p_b3, PRI, 128, BL_, 128, 200, 0, false);
    // 15-18) posterior MLP; layer 1 split over concat(H1, enc_o)  [edges]
    launch_gemm(stream, H1, H_, pQ1a, nullptr, T1, 200, BL_, 200, H_, 0, false);
    launch_gemm(stream, enc_o, E_, pQ1b, q_b1, T1, 200, BL_, 200, E_, 1, true);
    launch_gemm(stream, T1, 200, pQ2, q_b2, T2, 200, BL_, 200, 200, 1, false);
    launch_gemm(stream, T2, 200, pQ3, q_b3, POS, 128, BL_, 128, 200, 0, false);
    // 19) sampling heads
    finalize_k<<<SZ_ / 256, 256, 0, stream>>>(PRI, POS, noise_p, noise_q, out);
}
